// DenseDilatedKnnGraph_42082089566469
// MI455X (gfx1250) — matrix operand order repaired and checked by reading, not compiled, not run on hardware
//
#include <hip/hip_runtime.h>

#define N_PTS 10000
#define DIM   512
#define KNN   9
#define TILES (N_PTS / 16)   // 625
#define WPB   5              // waves per block (5*16 = 80 rows, 125 row-blocks)

typedef __attribute__((ext_vector_type(16))) __bf16 v16bf;
typedef __attribute__((ext_vector_type(8)))  __bf16 v8bf;
typedef __attribute__((ext_vector_type(8)))  float  v8f;
typedef unsigned int v4u __attribute__((ext_vector_type(4)));
typedef int          v8i __attribute__((ext_vector_type(8)));
typedef int          v4i __attribute__((ext_vector_type(4)));

union AFrag { v16bf v; v8bf h[2]; };

// ---------------- phase 1: column sum-of-squares (f32 global atomics) ----------
#define CS_ROWS 40
__global__ __launch_bounds__(256)
void colsq_kernel(const float* __restrict__ x, float* __restrict__ csum) {
  int t  = threadIdx.x;
  int r0 = blockIdx.x * CS_ROWS;
  float a0 = 0.f, a1 = 0.f;
  for (int r = 0; r < CS_ROWS; ++r) {
    const float* row = x + (size_t)(r0 + r) * DIM;
    float v0 = row[t];
    float v1 = row[t + 256];
    a0 += v0 * v0;
    a1 += v1 * v1;
  }
  atomicAdd(&csum[t], a0);
  atomicAdd(&csum[t + 256], a1);
}

// ------- phase 2: normalize columns, split into bf16 hi/lo, row sq-norm --------
__global__ __launch_bounds__(256)
void norm_kernel(const float* __restrict__ x, const float* __restrict__ csum,
                 __bf16* __restrict__ xh, __bf16* __restrict__ xl,
                 float* __restrict__ sq) {
  __shared__ float red[256];
  const int row = blockIdx.x, t = threadIdx.x;
  float local = 0.f;
  for (int c = t; c < DIM; c += 256) {
    float inv = 1.0f / fmaxf(sqrtf(csum[c]), 1e-12f);
    float v   = x[(size_t)row * DIM + c] * inv;
    local += v * v;
    __bf16 hi = (__bf16)v;
    xh[(size_t)row * DIM + c] = hi;
    xl[(size_t)row * DIM + c] = (__bf16)(v - (float)hi);
  }
  red[t] = local;
  __syncthreads();
  for (int s = 128; s > 0; s >>= 1) {
    if (t < s) red[t] += red[t + s];
    __syncthreads();
  }
  if (t == 0) sq[row] = red[0];
}

// ---- TDM descriptor: 2D tile, 16 rows x 512 bf16, row stride 512, to LDS ------
__device__ __forceinline__ void tdm_load_tile(const __bf16* gsrc, void* ldst) {
  unsigned long long ga = (unsigned long long)(uintptr_t)gsrc;
  unsigned int lds = (unsigned int)(uintptr_t)ldst;  // low 32 bits = LDS offset
  v4u g0;
  g0[0] = 1u;                                            // count=1, user D#
  g0[1] = lds;                                           // lds_addr
  g0[2] = (unsigned int)(ga & 0xFFFFFFFFu);              // global_addr[31:0]
  g0[3] = (unsigned int)((ga >> 32) & 0x01FFFFFFu)       // global_addr[56:32]
        | (2u << 30);                                    // type = 2 ("image")
  v8i g1;
  g1[0] = (int)(1u << 16);                               // data_size=1 (2 bytes)
  g1[1] = (int)((512u & 0xFFFFu) << 16);                 // tensor_dim0 lo16
  g1[2] = (int)(((512u >> 16) & 0xFFFFu) | (16u << 16)); // dim0 hi16 | dim1 lo16
  g1[3] = (int)(512u << 16);                             // dim1 hi16 | tile_dim0=512
  g1[4] = (int)16u;                                      // tile_dim1=16, tile_dim2=0
  g1[5] = (int)512;                                      // tensor_dim0_stride lo32
  g1[6] = 0;                                             // stride hi16 | dim1_stride lo
  g1[7] = 0;
  v4i g2 = {0, 0, 0, 0};
  v4i g3 = {0, 0, 0, 0};
  v8i g4 = {0, 0, 0, 0, 0, 0, 0, 0};   // clang-23 toolchain: 6-arg form
  __builtin_amdgcn_tensor_load_to_lds(g0, g1, g2, g3, g4, 0);
}

// ------- phase 3: TDM-staged split-bf16 WMMA GEMM + per-row running top-9 ------
// grid = 250: blockIdx>>1 selects 80 query rows, blockIdx&1 selects j-half.
__global__ __launch_bounds__(WPB * 32)
void knn_kernel(const __bf16* __restrict__ xh, const __bf16* __restrict__ xl,
                const float* __restrict__ sq,
                float* __restrict__ pV, int* __restrict__ pI) {
  __shared__ __bf16 btile[2][2][16][DIM];      // [buf][hi/lo][row][col] = 64 KB
  __shared__ float sc[WPB][16][17];
  __shared__ float bestV[WPB][16][KNN];
  __shared__ int   bestI[WPB][16][KNN];

  const int w    = threadIdx.x >> 5;
  const int lane = threadIdx.x & 31;
  const int m    = lane & 15;
  const int mh   = lane >> 4;
  const int half = blockIdx.x & 1;
  const int i0w  = (blockIdx.x >> 1) * (WPB * 16) + w * 16;  // this wave's rows

  for (int s = lane; s < 16 * KNN; s += 32) {
    bestV[w][s / KNN][s % KNN] = 3.4e38f;
    bestI[w][s / KNN][s % KNN] = 0;
  }

  // stage this wave's 16 A rows (hi+lo, full K) in registers
  const __bf16* Ah = xh + (size_t)(i0w + m) * DIM;
  const __bf16* Al = xl + (size_t)(i0w + m) * DIM;
  v16bf aH[16], aL[16];
#pragma unroll
  for (int kk = 0; kk < 16; ++kk) {
    const int k0 = kk * 32;
    AFrag t;
    t.h[0] = *(const v8bf*)(Ah + k0 + 16 * mh);
    t.h[1] = *(const v8bf*)(Ah + k0 + 16 * mh + 8);
    aH[kk] = t.v;
    t.h[0] = *(const v8bf*)(Al + k0 + 16 * mh);
    t.h[1] = *(const v8bf*)(Al + k0 + 16 * mh + 8);
    aL[kk] = t.v;
  }

  // prologue: TDM-load first B tile (hi+lo planes) into buffer 0
  int jt = half;
  if (w == 0) {
    tdm_load_tile(xh + (size_t)jt * 16 * DIM, &btile[0][0][0][0]);
    tdm_load_tile(xl + (size_t)jt * 16 * DIM, &btile[0][1][0][0]);
    __builtin_amdgcn_s_wait_tensorcnt(0);
  }
  __syncthreads();

  int buf = 0;
  for (; jt < TILES; jt += 2) {
    const int j0  = jt * 16;
    const int nxt = jt + 2;
    if (w == 0 && nxt < TILES) {            // async prefetch of next tile
      tdm_load_tile(xh + (size_t)nxt * 16 * DIM, &btile[buf ^ 1][0][0][0]);
      tdm_load_tile(xl + (size_t)nxt * 16 * DIM, &btile[buf ^ 1][1][0][0]);
    }

    v8f acc = {};
#pragma unroll
    for (int kk = 0; kk < 16; ++kk) {
      const int k0 = kk * 32;
      v16bf bh = *(const v16bf*)&btile[buf][0][m][k0 + 16 * mh];
      v16bf bl = *(const v16bf*)&btile[buf][1][m][k0 + 16 * mh];
      acc = __builtin_amdgcn_wmma_f32_16x16x32_bf16(false, aH[kk], false, bh,
                                                    (short)0, acc, false, false);
      acc = __builtin_amdgcn_wmma_f32_16x16x32_bf16(false, aH[kk], false, bl,
                                                    (short)0, acc, false, false);
      acc = __builtin_amdgcn_wmma_f32_16x16x32_bf16(false, aL[kk], false, bh,
                                                    (short)0, acc, false, false);
    }

    // C layout: VGPR r, lane -> (M = r + 8*mh, N = m). score = sq_j - 2*dot
    const float sqj = sq[j0 + m];
#pragma unroll
    for (int r = 0; r < 8; ++r)
      sc[w][r + 8 * mh][m] = __builtin_fmaf(-2.0f, acc[r], sqj);

    asm volatile("s_wait_dscnt 0" ::: "memory");   // intra-wave LDS RAW

    if (lane < 16) {                               // lane = row owner
      float* bv = bestV[w][lane];
      int*   bi = bestI[w][lane];
      for (int c = 0; c < 16; ++c) {
        float v = sc[w][lane][c];
        if (v < bv[KNN - 1]) {
          int p = KNN - 1;
          while (p > 0 && bv[p - 1] > v) {
            bv[p] = bv[p - 1]; bi[p] = bi[p - 1]; --p;
          }
          bv[p] = v; bi[p] = j0 + c;
        }
      }
    }

    if (w == 0 && nxt < TILES) __builtin_amdgcn_s_wait_tensorcnt(0);
    __syncthreads();                               // publish next buffer
    buf ^= 1;
  }

  // waves own disjoint rows: write this half's partial top-9 straight out
  if (lane < 16) {
    const int gi = i0w + lane;
    for (int s = 0; s < KNN; ++s) {
      pV[((size_t)half * N_PTS + gi) * KNN + s] = bestV[w][lane][s];
      pI[((size_t)half * N_PTS + gi) * KNN + s] = bestI[w][lane][s];
    }
  }
}

// ---------------- phase 4: merge the two j-half top-9 lists per row ------------
__global__ __launch_bounds__(256)
void merge_kernel(const float* __restrict__ pV, const int* __restrict__ pI,
                  int* __restrict__ out) {
  const int gi = blockIdx.x * 256 + threadIdx.x;
  if (gi >= N_PTS) return;
  float fv[KNN]; int fi[KNN];
  for (int s = 0; s < KNN; ++s) { fv[s] = 3.4e38f; fi[s] = 0; }
  for (int h = 0; h < 2; ++h)
    for (int s = 0; s < KNN; ++s) {
      float v = pV[((size_t)h * N_PTS + gi) * KNN + s];
      int idx = pI[((size_t)h * N_PTS + gi) * KNN + s];
      if (v < fv[KNN - 1]) {
        int p = KNN - 1;
        while (p > 0 && fv[p - 1] > v) {
          fv[p] = fv[p - 1]; fi[p] = fi[p - 1]; --p;
        }
        fv[p] = v; fi[p] = idx;
      }
    }
  for (int s = 0; s < KNN; ++s) {
    out[(size_t)gi * KNN + s] = fi[s];                       // nn_idx
    out[(size_t)N_PTS * KNN + (size_t)gi * KNN + s] = gi;    // center_idx
  }
}

extern "C" void kernel_launch(void* const* d_in, const int* in_sizes, int n_in,
                              void* d_out, int out_size, void* d_ws, size_t ws_size,
                              hipStream_t stream) {
  (void)in_sizes; (void)n_in; (void)out_size; (void)ws_size;
  const float* x = (const float*)d_in[0];   // [10000, 512] f32; d_in[1] = k (==9)

  char* ws = (char*)d_ws;
  float*  csum = (float*)ws;                                    //   512 f32
  float*  sq   = (float*)(ws + 4096);                           // 10000 f32
  float*  pV   = (float*)(ws + 65536);                          // 2*10000*9 f32
  int*    pI   = (int*)(ws + 65536 + 786432);                   // 2*10000*9 i32
  __bf16* xhp  = (__bf16*)(ws + 2097152);                       // bf16 hi plane
  __bf16* xlp  = (__bf16*)(ws + 2097152 + (size_t)N_PTS * DIM * 2); // bf16 lo
  int* out = (int*)d_out;

  (void)hipMemsetAsync(csum, 0, 512 * sizeof(float), stream);
  colsq_kernel<<<N_PTS / CS_ROWS, 256, 0, stream>>>(x, csum);
  norm_kernel<<<N_PTS, 256, 0, stream>>>(x, csum, xhp, xlp, sq);
  knn_kernel<<<250, WPB * 32, 0, stream>>>(xhp, xlp, sq, pV, pI);
  merge_kernel<<<(N_PTS + 255) / 256, 256, 0, stream>>>(pV, pI, out);
}
